// PointNet2Segmentation_43894565765763
// MI455X (gfx1250) — matrix operand order repaired and checked by reading, not compiled, not run on hardware
//
#include <hip/hip_runtime.h>

// ---------------------------------------------------------------------------
// PointNet++ segmentation forward for MI455X (gfx1250).
//
// Compute plan:
//  * All MLP layers run as f16 WMMA GEMMs (v_wmma_f32_16x16x32_f16, f32 acc)
//    with fused bias + eval-BatchNorm + ReLU epilogue. Weights are repacked
//    on-device into WMMA B-fragment order (one contiguous v16h per lane).
//  * Each wave owns NT (up to 4) adjacent 16-col output tiles: one A fragment
//    feeds NT WMMAs per K-step (4x WMMA density per LDS round-trip).
//  * A-tiles double-buffered through LDS (one barrier per K-step), staged in
//    the ISA-documented 16-bit A layout so each lane builds its fragment with
//    two 16B ds reads.
//  * FPS / ball-query / kNN-interp are small latency-bound kernels.
//
// Input flattening assumption (setup_inputs insertion order, each MLP dict
// flattened as W[0..], b[0..], g[0..], be[0..]):
//   0 feat, 1 pos,
//   sa1: W 2-4, b 5-7, g 8-9, be 10-11
//   sa2: W 12-14, b 15-17, g 18-19, be 20-21
//   sa3: W 22-24, b 25-27, g 28-29, be 30-31
//   fp3: W 32-33, b 34-35, g 36, be 37
//   fp2: W 38-39, b 40-41, g 42, be 43
//   fp1: W 44-46, b 47-49, g 50-51, be 52-53
//   head: W 54-56, b 57-59
// ---------------------------------------------------------------------------

typedef __attribute__((ext_vector_type(16))) _Float16 v16h;
typedef __attribute__((ext_vector_type(8)))  _Float16 v8h;
typedef __attribute__((ext_vector_type(8)))  float    v8f;

#define BATCH 16
#define NPTS  4096
#define KNBR  64
#define NS1   820
#define NS2   205
#define BN_SCALE 0.99999500003749975f  // 1/sqrt(1+1e-5)

static inline int cdiv_h(long a, long b) { return (int)((a + b - 1) / b); }

// ---------------------------------------------------------------------------
// Farthest point sampling: one block per batch, sequential greedy selection.
// ---------------------------------------------------------------------------
__global__ __launch_bounds__(1024) void fps_kernel(const float* __restrict__ pos,
                                                   int* __restrict__ out,
                                                   int Np, int ns) {
  const int b = blockIdx.x;
  const int tid = threadIdx.x;
  const float* P = pos + (size_t)b * Np * 3;
  __shared__ float sv[1024];
  __shared__ int   si[1024];
  __shared__ int   s_last;
  float mind[4];
  for (int j = 0; j < 4; ++j) mind[j] = 1e10f;
  if (tid == 0) { out[b * ns] = 0; s_last = 0; }
  __syncthreads();
  for (int it = 1; it < ns; ++it) {
    const int last = s_last;
    const float lx = P[last * 3 + 0], ly = P[last * 3 + 1], lz = P[last * 3 + 2];
    float bv = -1.0f; int bi = 0x7fffffff;
    for (int j = 0; j < 4; ++j) {
      const int p = tid + j * 1024;
      if (p < Np) {
        const float dx = P[p * 3 + 0] - lx;
        const float dy = P[p * 3 + 1] - ly;
        const float dz = P[p * 3 + 2] - lz;
        const float d = dx * dx + dy * dy + dz * dz;
        const float m = fminf(mind[j], d);
        mind[j] = m;
        if (m > bv || (m == bv && p < bi)) { bv = m; bi = p; }
      }
    }
    sv[tid] = bv; si[tid] = bi;
    __syncthreads();
    for (int s = 512; s > 0; s >>= 1) {
      if (tid < s) {
        const float v2 = sv[tid + s]; const int i2 = si[tid + s];
        if (v2 > sv[tid] || (v2 == sv[tid] && i2 < si[tid])) { sv[tid] = v2; si[tid] = i2; }
      }
      __syncthreads();
    }
    if (tid == 0) { s_last = si[0]; out[b * ns + it] = si[0]; }
    __syncthreads();
  }
}

__global__ void gather_centers(const float* __restrict__ pos, const int* __restrict__ cidx,
                               float* __restrict__ q, int Np, int ns) {
  const int t = blockIdx.x * blockDim.x + threadIdx.x;
  const int total = BATCH * ns * 3;
  if (t >= total) return;
  const int d = t % 3;
  const int ci = t / 3;
  const int b = ci / ns;
  q[t] = pos[((size_t)b * Np + cidx[ci]) * 3 + d];
}

// One 64-thread block per center; append-within-radius capped at KNBR.
__global__ __launch_bounds__(64) void ball_query_kernel(const float* __restrict__ q,
                                                        const float* __restrict__ pos,
                                                        int* __restrict__ nidx,
                                                        int* __restrict__ cnt,
                                                        int Np, int ns, float r2) {
  const int c = blockIdx.x;          // b*ns + i
  const int b = c / ns;
  const int tid = threadIdx.x;
  __shared__ int scnt;
  __shared__ int list[KNBR];
  if (tid == 0) scnt = 0;
  __syncthreads();
  const float qx = q[c * 3 + 0], qy = q[c * 3 + 1], qz = q[c * 3 + 2];
  const float* P = pos + (size_t)b * Np * 3;
  for (int p = tid; p < Np; p += 64) {
    const float dx = P[p * 3 + 0] - qx;
    const float dy = P[p * 3 + 1] - qy;
    const float dz = P[p * 3 + 2] - qz;
    if (dx * dx + dy * dy + dz * dz <= r2) {
      const int s = atomicAdd(&scnt, 1);
      if (s < KNBR) list[s] = p;
    }
  }
  __syncthreads();
  const int n = scnt < KNBR ? scnt : KNBR;
  if (tid == 0) cnt[c] = n;
  nidx[(size_t)c * KNBR + tid] = (tid < n) ? list[tid] : -1;
}

// Build padded f16 GEMM input rows: [x_j (C feats), pos_j - pos_i (3)], zero pad.
__global__ void build_sa_input(const float* __restrict__ xf32,
                               const _Float16* __restrict__ xf16, int C,
                               const float* __restrict__ pos, const float* __restrict__ q,
                               const int* __restrict__ nidx, _Float16* __restrict__ A,
                               int Np, int ns, int colsP) {
  const long total = (long)BATCH * ns * KNBR * colsP;
  const long stride = (long)gridDim.x * blockDim.x;
  for (long t = blockIdx.x * (long)blockDim.x + threadIdx.x; t < total; t += stride) {
    const int col = (int)(t % colsP);
    const long row = t / colsP;
    const int j = (int)(row % KNBR);
    const long ci = row / KNBR;
    const int b = (int)(ci / ns);
    float v = 0.0f;
    const int nb = nidx[ci * KNBR + j];
    if (nb >= 0) {
      if (col < C) {
        v = xf32 ? xf32[((size_t)b * Np + nb) * C + col]
                 : (float)xf16[((size_t)b * Np + nb) * C + col];
      } else if (col < C + 3) {
        const int d = col - C;
        v = pos[((size_t)b * Np + nb) * 3 + d] - q[ci * 3 + d];
      }
    }
    A[t] = (_Float16)v;
  }
}

// Repack f32 row-major W[K,N] into WMMA B-fragments: fragment (kt,nt) holds
// 32 lanes x 16 contiguous halfs, lane l / half v -> k = kt*32+8*(l/16)+v (v<8), kt*32+16+8*(l/16)+v-8 (v>=8),
// n = nt*16 + (l&15). Zero pad outside K,N.
__global__ void wrepack_kernel(const float* __restrict__ W, _Float16* __restrict__ F,
                               int Kd, int Nw) {
  const int ktiles = (Kd + 31) >> 5;
  const int ntiles = (Nw + 15) >> 4;
  const long total = (long)ktiles * ntiles * 512;
  const long stride = (long)gridDim.x * blockDim.x;
  for (long t = blockIdx.x * (long)blockDim.x + threadIdx.x; t < total; t += stride) {
    const int v = (int)(t & 15);
    const int l = (int)((t >> 4) & 31);
    const long ft = t >> 9;
    const int nt = (int)(ft % ntiles);
    const int kt = (int)(ft / ntiles);
    const int k = kt * 32 + (v < 8 ? 8 * (l >> 4) + v : 16 + 8 * (l >> 4) + (v - 8));
    const int n = nt * 16 + (l & 15);
    const float val = (k < Kd && n < Nw) ? W[(size_t)k * Nw + n] : 0.0f;
    F[t] = (_Float16)val;
  }
}

// ---------------------------------------------------------------------------
// WMMA GEMM: C[M,Nw] = act(A[M,Kp] @ W + bias). A is f16 row-major with row
// stride Kp = roundup32(Kd) (pad zeros). 8 waves/block; wave w -> rows
// block*128 + w*16, cols (blockIdx.y*NT .. +NT-1)*16. One A fragment feeds NT
// WMMAs; A tile double-buffered in LDS (one barrier per K-step).
// mode: 0 plain, 1 BN+ReLU, 2 ReLU.
// ---------------------------------------------------------------------------
template <int NT>
__global__ __launch_bounds__(256) void gemm_f16_wmma(
    const _Float16* __restrict__ A, const _Float16* __restrict__ Wf,
    const float* __restrict__ bias, const float* __restrict__ gamma,
    const float* __restrict__ beta,
    _Float16* __restrict__ out16, float* __restrict__ out32,
    int M, int Kd, int Nw, int mode) {
  __shared__ _Float16 lds[2][128 * 32];
  const int tid = threadIdx.x;
  const int lane = tid & 31;
  const int wv = tid >> 5;
  const int rowBase = blockIdx.x * 128;
  const int ntBase = blockIdx.y * NT;
  const int ktiles = (Kd + 31) >> 5;
  const int ntiles = (Nw + 15) >> 4;
  const int Kp = ktiles << 5;

  v8f acc[NT] = {};
  const int r = tid >> 1, seg = tid & 1;
  const long gm = rowBase + r;
  const int mrow = wv * 16 + (lane & 15);
  const int cofs = (lane >> 4) * 8;

  // Stage A tile kt into LDS buffer `buf` (row-major, 32-half rows).
  auto stage = [&](int kt, int buf) {
    v8h a0 = {}, a1 = {};
    if (gm < M) {
      const _Float16* src = A + gm * (long)Kp + kt * 32 + seg * 16;
      a0 = *(const v8h*)src;
      a1 = *(const v8h*)(src + 8);
    }
    *(v8h*)&lds[buf][r * 32 + seg * 16]     = a0;
    *(v8h*)&lds[buf][r * 32 + seg * 16 + 8] = a1;
  };

  stage(0, 0);
  int cur = 0;
  for (int kt = 0; kt < ktiles; ++kt) {
    __syncthreads();                 // buf[cur] staged; prior reads of buf[cur^1] done
    if (kt + 1 < ktiles) stage(kt + 1, cur ^ 1);

    // A fragment per ISA 16-bit A layout: lane group selects K halves.
    union { v16h v; v8h h[2]; } af;
    const _Float16* base = &lds[cur][mrow * 32];
    af.h[0] = *(const v8h*)(base + cofs);        // K = c..c+7
    af.h[1] = *(const v8h*)(base + cofs + 16);   // K = c+16..c+23

#pragma unroll
    for (int t = 0; t < NT; ++t) {
      // B fragment: one contiguous 32B per lane from the repacked buffer.
      const v16h bf = *(const v16h*)&Wf[(((long)kt * ntiles + ntBase + t) * 32 + lane) * 16];
      acc[t] = __builtin_amdgcn_wmma_f32_16x16x32_f16(false, af.v, false, bf,
                                                      (short)0, acc[t], false, false);
    }
    cur ^= 1;
  }

  // Epilogue: bias + optional eval-BN + ReLU; C layout: VGPR r -> M=r+8*(lane/16).
#pragma unroll
  for (int t = 0; t < NT; ++t) {
    const int ncol = (ntBase + t) * 16 + (lane & 15);
    if (ncol < Nw) {
      const float bb = bias[ncol];
      float gs = 0.f, be = 0.f;
      if (mode == 1) { gs = gamma[ncol] * BN_SCALE; be = beta[ncol]; }
      union { v8f v; float f[8]; } res; res.v = acc[t];
      for (int rr = 0; rr < 8; ++rr) {
        const long m = rowBase + wv * 16 + rr + 8 * (lane >> 4);
        if (m < M) {
          float v = res.f[rr] + bb;
          if (mode == 1) v = fmaxf(v * gs + be, 0.0f);
          else if (mode == 2) v = fmaxf(v, 0.0f);
          if (out32) out32[m * Nw + ncol] = v;
          else out16[m * (long)Nw + ncol] = (_Float16)v;
        }
      }
    }
  }
}

// Max over up to cnt neighbors (masked max-aggregation).
__global__ void max_aggregate(const _Float16* __restrict__ h, const int* __restrict__ cnt,
                              _Float16* __restrict__ out, int ns, int C) {
  const long total = (long)BATCH * ns * C;
  const long t = blockIdx.x * (long)blockDim.x + threadIdx.x;
  if (t >= total) return;
  const int ch = (int)(t % C);
  const long ci = t / C;
  const int n = cnt[ci];
  float m = -1e30f;
  for (int j = 0; j < n; ++j)
    m = fmaxf(m, (float)h[(ci * KNBR + j) * (long)C + ch]);
  out[t] = (_Float16)m;
}

__global__ void build_sa3_input(const _Float16* __restrict__ x2,
                                const float* __restrict__ pos2,
                                _Float16* __restrict__ A) {
  const int colsP = 288;                          // 259 padded to 32
  const long total = (long)BATCH * NS2 * colsP;
  const long t = blockIdx.x * (long)blockDim.x + threadIdx.x;
  if (t >= total) return;
  const int col = (int)(t % colsP);
  const long row = t / colsP;
  float v = 0.f;
  if (col < 256) v = (float)x2[row * 256 + col];
  else if (col < 259) v = pos2[row * 3 + (col - 256)];
  A[t] = (_Float16)v;
}

__global__ void global_max_points(const _Float16* __restrict__ h, _Float16* __restrict__ out,
                                  int npts, int C) {
  const int t = blockIdx.x * blockDim.x + threadIdx.x;
  if (t >= BATCH * C) return;
  const int c = t % C, b = t / C;
  float m = -1e30f;
  for (int i = 0; i < npts; ++i)
    m = fmaxf(m, (float)h[((size_t)b * npts + i) * C + c]);
  out[t] = (_Float16)m;
}

// fp3 input: [x3 broadcast (1024) | x2 (256)]  (knn k=1 from single source).
__global__ void build_fp3_input(const _Float16* __restrict__ x3,
                                const _Float16* __restrict__ x2,
                                _Float16* __restrict__ A) {
  const int cols = 1280;
  const long total = (long)BATCH * NS2 * cols;
  const long t = blockIdx.x * (long)blockDim.x + threadIdx.x;
  if (t >= total) return;
  const int col = (int)(t % cols);
  const long row = t / cols;
  const int b = (int)(row / NS2);
  A[t] = (col < 1024) ? x3[b * 1024 + col] : x2[row * 256 + (col - 1024)];
}

// 3-NN inverse-d^2 weighted interpolation; one block per destination point.
__global__ __launch_bounds__(128) void knn3_interp(
    const _Float16* __restrict__ xs, const float* __restrict__ ps,
    const float* __restrict__ pd, _Float16* __restrict__ out,
    int Ns, int Nd, int C) {
  const int g = blockIdx.x;
  const int b = g / Nd, i = g % Nd;
  const int tid = threadIdx.x;
  __shared__ float sd[128 * 3];
  __shared__ int   sidx[128 * 3];
  __shared__ float sw[3];
  __shared__ int   sbi[3];
  const float px = pd[((size_t)b * Nd + i) * 3 + 0];
  const float py = pd[((size_t)b * Nd + i) * 3 + 1];
  const float pz = pd[((size_t)b * Nd + i) * 3 + 2];
  float bd[3] = {1e30f, 1e30f, 1e30f}; int bi[3] = {0, 0, 0};
  const float* P = ps + (size_t)b * Ns * 3;
  for (int s = tid; s < Ns; s += 128) {
    const float dx = P[s * 3 + 0] - px;
    const float dy = P[s * 3 + 1] - py;
    const float dz = P[s * 3 + 2] - pz;
    const float d = dx * dx + dy * dy + dz * dz;
    if (d < bd[0])      { bd[2]=bd[1]; bi[2]=bi[1]; bd[1]=bd[0]; bi[1]=bi[0]; bd[0]=d; bi[0]=s; }
    else if (d < bd[1]) { bd[2]=bd[1]; bi[2]=bi[1]; bd[1]=d; bi[1]=s; }
    else if (d < bd[2]) { bd[2]=d; bi[2]=s; }
  }
  for (int k = 0; k < 3; ++k) { sd[tid * 3 + k] = bd[k]; sidx[tid * 3 + k] = bi[k]; }
  __syncthreads();
  if (tid == 0) {
    float gd[3] = {1e30f, 1e30f, 1e30f}; int gi[3] = {0, 0, 0};
    for (int s = 0; s < 128 * 3; ++s) {
      const float d = sd[s]; const int id = sidx[s];
      if (d < gd[0])      { gd[2]=gd[1]; gi[2]=gi[1]; gd[1]=gd[0]; gi[1]=gi[0]; gd[0]=d; gi[0]=id; }
      else if (d < gd[1]) { gd[2]=gd[1]; gi[2]=gi[1]; gd[1]=d; gi[1]=id; }
      else if (d < gd[2]) { gd[2]=d; gi[2]=id; }
    }
    float ws = 0.f, w[3];
    for (int k = 0; k < 3; ++k) { w[k] = 1.0f / fmaxf(gd[k], 1e-16f); ws += w[k]; }
    for (int k = 0; k < 3; ++k) { sw[k] = w[k] / ws; sbi[k] = gi[k]; }
  }
  __syncthreads();
  const float w0 = sw[0], w1 = sw[1], w2 = sw[2];
  const int i0 = sbi[0], i1 = sbi[1], i2 = sbi[2];
  const _Float16* X = xs + (size_t)b * Ns * C;
  for (int ch = tid; ch < C; ch += 128) {
    const float v = w0 * (float)X[(size_t)i0 * C + ch] +
                    w1 * (float)X[(size_t)i1 * C + ch] +
                    w2 * (float)X[(size_t)i2 * C + ch];
    out[((size_t)b * Nd + i) * C + ch] = (_Float16)v;
  }
}

__global__ void concat_ff(const _Float16* __restrict__ a, int Ca,
                          const _Float16* __restrict__ bsrc, int Cb,
                          _Float16* __restrict__ out, long rows, int colsP) {
  const long total = rows * colsP;
  const long t = blockIdx.x * (long)blockDim.x + threadIdx.x;
  if (t >= total) return;
  const int col = (int)(t % colsP);
  const long row = t / colsP;
  _Float16 v = (_Float16)0.f;
  if (col < Ca) v = a[row * Ca + col];
  else if (col < Ca + Cb) v = bsrc[row * Cb + (col - Ca)];
  out[t] = v;
}

__global__ void concat_f16_f32(const _Float16* __restrict__ a, int Ca,
                               const float* __restrict__ bsrc, int Cb,
                               _Float16* __restrict__ out, long rows, int colsP) {
  const long total = rows * colsP;
  const long t = blockIdx.x * (long)blockDim.x + threadIdx.x;
  if (t >= total) return;
  const int col = (int)(t % colsP);
  const long row = t / colsP;
  float v = 0.f;
  if (col < Ca) v = (float)a[row * Ca + col];
  else if (col < Ca + Cb) v = bsrc[row * Cb + (col - Ca)];
  out[t] = (_Float16)v;
}

// ---------------------------------------------------------------------------
// Host orchestration
// ---------------------------------------------------------------------------
extern "C" void kernel_launch(void* const* d_in, const int* in_sizes, int n_in,
                              void* d_out, int out_size, void* d_ws, size_t ws_size,
                              hipStream_t stream) {
  (void)in_sizes; (void)n_in; (void)out_size; (void)ws_size;
  const float* feat = (const float*)d_in[0];
  const float* pos  = (const float*)d_in[1];
  auto F = [&](int i) { return (const float*)d_in[i]; };

  // ---- workspace carving (deterministic) ----
  char* ws = (char*)d_ws;
  size_t cur = 0;
  auto alloc = [&](size_t bytes) -> char* {
    char* p = ws + cur;
    cur = (cur + bytes + 255) & ~(size_t)255;
    return p;
  };

  struct WI { int idx, K, N; };
  const WI wi[19] = {
    {2, 6, 64},   {3, 64, 64},   {4, 64, 128},      // sa1
    {12, 131, 128},{13, 128, 128},{14, 128, 256},   // sa2
    {22, 259, 256},{23, 256, 512},{24, 512, 1024},  // sa3
    {32, 1280, 256},{33, 256, 256},                 // fp3
    {38, 384, 256},{39, 256, 128},                  // fp2
    {44, 131, 128},{45, 128, 128},{46, 128, 128},   // fp1
    {54, 128, 128},{55, 128, 128},{56, 128, 13},    // head
  };
  _Float16* frag[19];
  long fragHalfs[19];
  for (int w = 0; w < 19; ++w) {
    fragHalfs[w] = (long)cdiv_h(wi[w].K, 32) * cdiv_h(wi[w].N, 16) * 512;
    frag[w] = (_Float16*)alloc((size_t)fragHalfs[w] * 2);
  }

  int*      cidx1 = (int*)alloc((size_t)BATCH * NS1 * 4);
  float*    pos1  = (float*)alloc((size_t)BATCH * NS1 * 3 * 4);
  int*      nidx1 = (int*)alloc((size_t)BATCH * NS1 * KNBR * 4);
  int*      cnt1  = (int*)alloc((size_t)BATCH * NS1 * 4);
  _Float16* x1    = (_Float16*)alloc((size_t)BATCH * NS1 * 128 * 2);
  int*      cidx2 = (int*)alloc((size_t)BATCH * NS2 * 4);
  float*    pos2  = (float*)alloc((size_t)BATCH * NS2 * 3 * 4);
  int*      nidx2 = (int*)alloc((size_t)BATCH * NS2 * KNBR * 4);
  int*      cnt2  = (int*)alloc((size_t)BATCH * NS2 * 4);
  _Float16* x2    = (_Float16*)alloc((size_t)BATCH * NS2 * 256 * 2);
  _Float16* x3    = (_Float16*)alloc((size_t)BATCH * 1024 * 2);
  _Float16* y2    = (_Float16*)alloc((size_t)BATCH * NS2 * 256 * 2);
  _Float16* y2i   = (_Float16*)alloc((size_t)BATCH * NS1 * 256 * 2);
  _Float16* y1    = (_Float16*)alloc((size_t)BATCH * NS1 * 128 * 2);
  _Float16* y1i   = (_Float16*)alloc((size_t)BATCH * NPTS * 128 * 2);
  const size_t PH = (size_t)BATCH * NS1 * KNBR * 128;     // largest chain buffer
  _Float16* ping  = (_Float16*)alloc(PH * 2);
  _Float16* pong  = (_Float16*)alloc(PH * 2);

  // ---- weight repack into WMMA B-fragment order ----
  for (int w = 0; w < 19; ++w) {
    int blocks = cdiv_h(fragHalfs[w], 256); if (blocks > 4096) blocks = 4096;
    wrepack_kernel<<<blocks, 256, 0, stream>>>(F(wi[w].idx), frag[w], wi[w].K, wi[w].N);
  }

  auto gemm = [&](const _Float16* A, int widx, int bIdx, int gIdx, int beIdx,
                  _Float16* o16, float* o32, int M, int mode) {
    const int Kd = wi[widx].K, Nw = wi[widx].N;
    const int ntiles = cdiv_h(Nw, 16);
    const float* gp = gIdx >= 0 ? F(gIdx) : nullptr;
    const float* bp = beIdx >= 0 ? F(beIdx) : nullptr;
    if ((ntiles & 3) == 0) {
      dim3 gr(cdiv_h(M, 128), ntiles / 4);
      gemm_f16_wmma<4><<<gr, 256, 0, stream>>>(A, frag[widx], F(bIdx), gp, bp,
                                               o16, o32, M, Kd, Nw, mode);
    } else if ((ntiles & 1) == 0) {
      dim3 gr(cdiv_h(M, 128), ntiles / 2);
      gemm_f16_wmma<2><<<gr, 256, 0, stream>>>(A, frag[widx], F(bIdx), gp, bp,
                                               o16, o32, M, Kd, Nw, mode);
    } else {
      dim3 gr(cdiv_h(M, 128), ntiles);
      gemm_f16_wmma<1><<<gr, 256, 0, stream>>>(A, frag[widx], F(bIdx), gp, bp,
                                               o16, o32, M, Kd, Nw, mode);
    }
  };

  const int MSA1 = BATCH * NS1 * KNBR;   // 839680
  const int MSA2 = BATCH * NS2 * KNBR;   // 209920
  const int M3   = BATCH * NS2;          // 3280
  const int Mfp2 = BATCH * NS1;          // 13120
  const int Mfp1 = BATCH * NPTS;         // 65536

  // ================= SA1 =================
  fps_kernel<<<BATCH, 1024, 0, stream>>>(pos, cidx1, NPTS, NS1);
  gather_centers<<<cdiv_h((long)BATCH * NS1 * 3, 256), 256, 0, stream>>>(pos, cidx1, pos1, NPTS, NS1);
  ball_query_kernel<<<BATCH * NS1, 64, 0, stream>>>(pos1, pos, nidx1, cnt1, NPTS, NS1, 0.2f * 0.2f);
  build_sa_input<<<4096, 256, 0, stream>>>(feat, nullptr, 3, pos, pos1, nidx1, ping, NPTS, NS1, 32);
  gemm(ping, 0, 5, 8, 10, pong, nullptr, MSA1, 1);        // 6->64 BN+ReLU
  gemm(pong, 1, 6, 9, 11, ping, nullptr, MSA1, 1);        // 64->64 BN+ReLU
  gemm(ping, 2, 7, -1, -1, pong, nullptr, MSA1, 0);       // 64->128 plain
  max_aggregate<<<cdiv_h((long)BATCH * NS1 * 128, 256), 256, 0, stream>>>(pong, cnt1, x1, NS1, 128);

  // ================= SA2 =================
  fps_kernel<<<BATCH, 1024, 0, stream>>>(pos1, cidx2, NS1, NS2);
  gather_centers<<<cdiv_h((long)BATCH * NS2 * 3, 256), 256, 0, stream>>>(pos1, cidx2, pos2, NS1, NS2);
  ball_query_kernel<<<BATCH * NS2, 64, 0, stream>>>(pos2, pos1, nidx2, cnt2, NS1, NS2, 0.4f * 0.4f);
  build_sa_input<<<4096, 256, 0, stream>>>(nullptr, x1, 128, pos1, pos2, nidx2, ping, NS1, NS2, 160);
  gemm(ping, 3, 15, 18, 20, pong, nullptr, MSA2, 1);      // 131->128
  gemm(pong, 4, 16, 19, 21, ping, nullptr, MSA2, 1);      // 128->128
  gemm(ping, 5, 17, -1, -1, pong, nullptr, MSA2, 0);      // 128->256 plain
  max_aggregate<<<cdiv_h((long)BATCH * NS2 * 256, 256), 256, 0, stream>>>(pong, cnt2, x2, NS2, 256);

  // ================= SA3 (global) =================
  build_sa3_input<<<cdiv_h((long)M3 * 288, 256), 256, 0, stream>>>(x2, pos2, ping);
  gemm(ping, 6, 25, 28, 30, pong, nullptr, M3, 1);        // 259->256
  gemm(pong, 7, 26, 29, 31, ping, nullptr, M3, 1);        // 256->512
  gemm(ping, 8, 27, -1, -1, pong, nullptr, M3, 0);        // 512->1024 plain
  global_max_points<<<cdiv_h(BATCH * 1024, 256), 256, 0, stream>>>(pong, x3, NS2, 1024);

  // ================= FP3 =================
  build_fp3_input<<<cdiv_h((long)M3 * 1280, 256), 256, 0, stream>>>(x3, x2, ping);
  gemm(ping, 9, 34, 36, 37, pong, nullptr, M3, 1);        // 1280->256
  gemm(pong, 10, 35, -1, -1, y2, nullptr, M3, 0);         // 256->256 plain
  knn3_interp<<<BATCH * NS1, 128, 0, stream>>>(y2, pos2, pos1, y2i, NS2, NS1, 256);

  // ================= FP2 =================
  concat_ff<<<cdiv_h((long)Mfp2 * 384, 256), 256, 0, stream>>>(y2i, 256, x1, 128, ping, Mfp2, 384);
  gemm(ping, 11, 40, 42, 43, pong, nullptr, Mfp2, 1);     // 384->256
  gemm(pong, 12, 41, -1, -1, y1, nullptr, Mfp2, 0);       // 256->128 plain
  knn3_interp<<<BATCH * NPTS, 128, 0, stream>>>(y1, pos1, pos, y1i, NS1, NPTS, 128);

  // ================= FP1 =================
  concat_f16_f32<<<cdiv_h((long)Mfp1 * 160, 256), 256, 0, stream>>>(y1i, 128, feat, 3, ping, Mfp1, 160);
  gemm(ping, 13, 47, 50, 52, pong, nullptr, Mfp1, 1);     // 131->128
  gemm(pong, 14, 48, 51, 53, ping, nullptr, Mfp1, 1);     // 128->128
  gemm(ping, 15, 49, -1, -1, pong, nullptr, Mfp1, 0);     // 128->128 plain

  // ================= HEAD (norm=False) =================
  gemm(pong, 16, 57, -1, -1, ping, nullptr, Mfp1, 2);     // 128->128 ReLU
  gemm(ping, 17, 58, -1, -1, pong, nullptr, Mfp1, 2);     // 128->128 ReLU
  gemm(pong, 18, 59, -1, -1, nullptr, (float*)d_out, Mfp1, 0);  // 128->13 -> f32 out
}
